// MultiScaleRetentionTriton_16441134809483
// MI455X (gfx1250) — compile-verified, matrix operand order checked by reading, not run on hardware
//
#include <hip/hip_runtime.h>

// ---------------------------------------------------------------------------
// MultiScaleRetention on MI455X (gfx1250, wave32, WMMA, async-LDS DMA)
//   q,v,g = x @ {Wq,Wv,Wg}^T   (bf16 WMMA, f32 accum)
//   state = decay-scan(v)       (chunked 3-phase scan)
//   y = LN(q*state) * silu(g)   (fused row kernel, fixup folded in)
//   out = y @ Wo^T              (bf16 WMMA, f32 out)
// ---------------------------------------------------------------------------

typedef unsigned short u16;
typedef __attribute__((ext_vector_type(16))) __bf16 v16bf;
typedef __attribute__((ext_vector_type(8)))  float  v8f;

#define D_MODEL 1024
#define NHEAD   16
#define HEAD_DIM 64
#define BATCH   4
#define SEQ     8192
#define CLEN    512
#define NCHUNK  (SEQ / CLEN)

__device__ __forceinline__ u16 f32_to_bf16(float f) {
  unsigned int u = __float_as_uint(f);
  unsigned int r = (u + 0x7FFFu + ((u >> 16) & 1u)) >> 16;   // RNE
  return (u16)r;
}
__device__ __forceinline__ float bf16_to_f32(u16 h) {
  return __uint_as_float(((unsigned int)h) << 16);
}

// LDS byte offset of a __shared__ object: low 32 bits of the flat address
// (per gfx1250 aperture rules).
__device__ __forceinline__ unsigned int lds_offset(const void* p) {
  return (unsigned int)(unsigned long long)(uintptr_t)p;
}

// Async global -> LDS DMA, 16 bytes per lane, tracked by ASYNCcnt.
__device__ __forceinline__ void async_copy_b128(unsigned int lds_off,
                                                const void* gaddr) {
  asm volatile("global_load_async_to_lds_b128 %0, %1, off"
               :: "v"(lds_off),
                  "v"((unsigned long long)(uintptr_t)gaddr)
               : "memory");
}
__device__ __forceinline__ void wait_async0() {
  asm volatile("s_wait_asynccnt 0x0" ::: "memory");
}

// ------------------------------- casts -------------------------------------
__global__ void cast_f32_to_bf16_kernel(const float* __restrict__ in,
                                        u16* __restrict__ out, size_t n) {
  size_t i = (size_t)blockIdx.x * blockDim.x + threadIdx.x;
  if (i < n) out[i] = f32_to_bf16(in[i]);
}

// ------------------------------- GEMM --------------------------------------
// Out[m][n] = sum_k A[m][k] * W[n][k]   (A: MxK bf16 row-major, W: NxK bf16)
#define BM 128
#define BN 128
#define BK 32
#define LDSS 40   // padded row stride (u16) -> 80B rows, conflict-spread

union Frag16 { v16bf v; uint4 q[2]; };

// ISA 16-bit A/B layout (16x32): lane half h, element i:
//   i in [0,8):  K = h*8 + i          -> one contiguous b128
//   i in [8,16): K = 16 + h*8 + (i-8) -> one contiguous b128
__device__ __forceinline__ v16bf load_frag(const u16* row, int half) {
  Frag16 f;
  f.q[0] = *(const uint4*)(row + half * 8);
  f.q[1] = *(const uint4*)(row + 16 + half * 8);
  return f.v;
}

template <int OUT_BF16>
__global__ __launch_bounds__(256)
void gemm_xwT_kernel(const u16* __restrict__ A, const u16* __restrict__ W,
                     void* __restrict__ Out, int M, int N, int K) {
  // double-buffered tiles, filled by async DMA (no VGPR staging)
  __shared__ __align__(16) u16 As[2][BM * LDSS];
  __shared__ __align__(16) u16 Bs[2][BN * LDSS];

  const int tid  = threadIdx.x;
  const int lane = tid & 31;
  const int wave = tid >> 5;
  const int half = lane >> 4;
  const int l16  = lane & 15;
  const int waveM = wave & 1;   // 2 waves along M
  const int waveN = wave >> 1;  // 4 waves along N

  const int blockM = blockIdx.y * BM;
  const int blockN = blockIdx.x * BN;

  // each thread DMAs 16 bf16 of A and 16 of B per K-step (2x b128 each)
  const int srow = tid >> 1;               // 0..127
  const int skg  = (tid & 1) * 16;         // 0 or 16

  const u16* aPtr = A + (size_t)(blockM + srow) * K + skg;
  const u16* bPtr = W + (size_t)(blockN + srow) * K + skg;

  const unsigned int ldsA0 = lds_offset(&As[0][srow * LDSS + skg]);
  const unsigned int ldsA1 = lds_offset(&As[1][srow * LDSS + skg]);
  const unsigned int ldsB0 = lds_offset(&Bs[0][srow * LDSS + skg]);
  const unsigned int ldsB1 = lds_offset(&Bs[1][srow * LDSS + skg]);

  v8f acc[4][2];
  v8f vz = {};
#pragma unroll
  for (int i = 0; i < 4; i++)
#pragma unroll
    for (int j = 0; j < 2; j++) acc[i][j] = vz;

  // kick off DMA of tile 0 into buffer 0
  async_copy_b128(ldsA0, aPtr);
  async_copy_b128(ldsA0 + 16, aPtr + 8);
  async_copy_b128(ldsB0, bPtr);
  async_copy_b128(ldsB0 + 16, bPtr + 8);

  int buf = 0;
  for (int k0 = 0; k0 < K; k0 += BK) {
    wait_async0();                // my DMA writes for this buffer landed
    __syncthreads();              // everyone's DMA writes visible

    if (k0 + BK < K) {            // DMA next tile into other buffer,
      aPtr += BK;                 // overlapped with the WMMA below
      bPtr += BK;
      const unsigned int la = buf ? ldsA0 : ldsA1;
      const unsigned int lb = buf ? ldsB0 : ldsB1;
      async_copy_b128(la, aPtr);
      async_copy_b128(la + 16, aPtr + 8);
      async_copy_b128(lb, bPtr);
      async_copy_b128(lb + 16, bPtr + 8);
    }

    const u16* aT = As[buf];
    const u16* bT = Bs[buf];
    v16bf af[4], bfr[2];
#pragma unroll
    for (int mt = 0; mt < 4; mt++)
      af[mt] = load_frag(aT + (waveM * 64 + mt * 16 + l16) * LDSS, half);
#pragma unroll
    for (int nt = 0; nt < 2; nt++)
      bfr[nt] = load_frag(bT + (waveN * 32 + nt * 16 + l16) * LDSS, half);

#pragma unroll
    for (int mt = 0; mt < 4; mt++)
#pragma unroll
      for (int nt = 0; nt < 2; nt++)
        acc[mt][nt] = __builtin_amdgcn_wmma_f32_16x16x32_bf16(
            false, af[mt], false, bfr[nt], (short)0, acc[mt][nt], false, false);

    buf ^= 1;
  }

  // epilogue: C VGPR r -> row = base + half*8 + r, col = base + l16
#pragma unroll
  for (int mt = 0; mt < 4; mt++) {
#pragma unroll
    for (int nt = 0; nt < 2; nt++) {
      const int col  = blockN + waveN * 32 + nt * 16 + l16;
      const int rowb = blockM + waveM * 64 + mt * 16 + half * 8;
      v8f c = acc[mt][nt];
      if (OUT_BF16) {
        u16* o = (u16*)Out;
#pragma unroll
        for (int r = 0; r < 8; r++)
          o[(size_t)(rowb + r) * N + col] = f32_to_bf16(c[r]);
      } else {
        float* o = (float*)Out;
#pragma unroll
        for (int r = 0; r < 8; r++)
          o[(size_t)(rowb + r) * N + col] = c[r];
      }
    }
  }
}

// --------------------------- decay scan ------------------------------------
// Pass 1: local inclusive scan per (b, channel, chunk), in place; emit carry.
__global__ __launch_bounds__(256)
void scan_chunk_kernel(float* __restrict__ v, const float* __restrict__ decay,
                       float* __restrict__ carry) {
  const int g     = blockIdx.x * blockDim.x + threadIdx.x;
  const int chunk = g / (BATCH * D_MODEL);
  const int rem   = g % (BATCH * D_MODEL);
  const int b     = rem / D_MODEL;
  const int c     = rem % D_MODEL;          // h*64 + d
  const float lam = decay[c >> 6];

  size_t base = (size_t)b * SEQ * D_MODEL + (size_t)chunk * CLEN * D_MODEL + c;
  float s = 0.f;
  for (int t = 0; t < CLEN; t++) {
    s = lam * s + v[base];
    v[base] = s;
    base += D_MODEL;
  }
  carry[((size_t)(b * D_MODEL + c)) * NCHUNK + chunk] = s;
}

// Pass 2: in-place exclusive combine across chunks; carry[j] becomes the
// state to inject at the start of chunk j.
__global__ __launch_bounds__(256)
void scan_carry_kernel(float* __restrict__ carry,
                       const float* __restrict__ decay) {
  const int g = blockIdx.x * blockDim.x + threadIdx.x;  // B*D_MODEL threads
  const int c = g % D_MODEL;
  const float lamP = __powf(decay[c >> 6], (float)CLEN);
  float* p = carry + (size_t)g * NCHUNK;
  float E = 0.f;
  for (int j = 0; j < NCHUNK; j++) {
    float e = p[j];
    p[j] = E;
    E = e + lamP * E;
  }
}

// Pass 3 fused with LN + SiLU gate: one block per (b,t) row of 1024.
__global__ __launch_bounds__(256)
void ln_gate_kernel(const float* __restrict__ vstate, const u16* __restrict__ qb,
                    const u16* __restrict__ gb, const float* __restrict__ carry,
                    const float* __restrict__ decay,
                    const float* __restrict__ gamma,
                    const float* __restrict__ beta, u16* __restrict__ yg) {
  const int row   = blockIdx.x;           // b*SEQ + t
  const int b     = row / SEQ;
  const int t     = row % SEQ;
  const int chunk = t / CLEN;
  const int tl    = t % CLEN;
  const int tid   = threadIdx.x;
  const size_t rbase = (size_t)row * D_MODEL;

  const int e0 = tid * 4;                 // 4 contiguous elems, same head
  const float lam = decay[e0 >> 6];
  const float pw  = __powf(lam, (float)(tl + 1));

  float4 vl = *(const float4*)(vstate + rbase + e0);
  const float* cp = carry + ((size_t)(b * D_MODEL + e0)) * NCHUNK + chunk;
  float st[4];
  st[0] = vl.x + pw * cp[0];
  st[1] = vl.y + pw * cp[NCHUNK];
  st[2] = vl.z + pw * cp[2 * NCHUNK];
  st[3] = vl.w + pw * cp[3 * NCHUNK];

  const u16* qp = qb + rbase + e0;
  float y[4];
  float sum = 0.f, sum2 = 0.f;
#pragma unroll
  for (int j = 0; j < 4; j++) {
    y[j] = bf16_to_f32(qp[j]) * st[j];
    sum += y[j];
    sum2 += y[j] * y[j];
  }

  __shared__ float red1[256];
  __shared__ float red2[256];
  __shared__ float s_mu, s_rsig;
  red1[tid] = sum;
  red2[tid] = sum2;
  __syncthreads();
#pragma unroll
  for (int s = 128; s > 0; s >>= 1) {
    if (tid < s) {
      red1[tid] += red1[tid + s];
      red2[tid] += red2[tid + s];
    }
    __syncthreads();
  }
  if (tid == 0) {
    float mu  = red1[0] * (1.f / D_MODEL);
    float var = red2[0] * (1.f / D_MODEL) - mu * mu;
    s_mu = mu;
    s_rsig = rsqrtf(var + 1e-5f);
  }
  __syncthreads();
  const float mu = s_mu, rsig = s_rsig;

  const u16* gp = gb + rbase + e0;
#pragma unroll
  for (int j = 0; j < 4; j++) {
    float gvl = bf16_to_f32(gp[j]);
    float sg  = gvl / (1.f + __expf(-gvl));            // silu
    float o = ((y[j] - mu) * rsig * gamma[e0 + j] + beta[e0 + j]) * sg;
    yg[rbase + e0 + j] = f32_to_bf16(o);
  }
}

// ------------------------------ launch -------------------------------------
extern "C" void kernel_launch(void* const* d_in, const int* in_sizes, int n_in,
                              void* d_out, int out_size, void* d_ws,
                              size_t ws_size, hipStream_t stream) {
  const float* x     = (const float*)d_in[0];
  const float* decay = (const float*)d_in[1];
  const float* Wq    = (const float*)d_in[2];
  const float* Wv    = (const float*)d_in[3];
  const float* Wg    = (const float*)d_in[4];
  const float* Wo    = (const float*)d_in[5];
  const float* gamma = (const float*)d_in[6];
  const float* beta  = (const float*)d_in[7];

  const size_t ML = (size_t)BATCH * SEQ;          // 32768 rows
  const size_t nXD = ML * D_MODEL;                // 33.5M elems
  const size_t nWD = (size_t)D_MODEL * D_MODEL;   // 1M elems

  char* ws = (char*)d_ws;
  size_t off = 0;
  u16* xb  = (u16*)(ws + off); off += nXD * sizeof(u16);  // also reused as yg
  u16* Wqb = (u16*)(ws + off); off += nWD * sizeof(u16);
  u16* Wvb = (u16*)(ws + off); off += nWD * sizeof(u16);
  u16* Wgb = (u16*)(ws + off); off += nWD * sizeof(u16);
  u16* Wob = (u16*)(ws + off); off += nWD * sizeof(u16);
  u16* qb  = (u16*)(ws + off); off += nXD * sizeof(u16);
  float* vst = (float*)(ws + off); off += nXD * sizeof(float);
  u16* gb  = (u16*)(ws + off); off += nXD * sizeof(u16);
  float* carry = (float*)(ws + off);
  off += (size_t)BATCH * D_MODEL * NCHUNK * sizeof(float);
  (void)off; (void)ws_size; (void)in_sizes; (void)n_in; (void)out_size;

  // ---- precision prep: fp32 -> bf16 (GEMM operands) ----
  cast_f32_to_bf16_kernel<<<(unsigned)(nXD / 256), 256, 0, stream>>>(x, xb, nXD);
  cast_f32_to_bf16_kernel<<<(unsigned)(nWD / 256), 256, 0, stream>>>(Wq, Wqb, nWD);
  cast_f32_to_bf16_kernel<<<(unsigned)(nWD / 256), 256, 0, stream>>>(Wv, Wvb, nWD);
  cast_f32_to_bf16_kernel<<<(unsigned)(nWD / 256), 256, 0, stream>>>(Wg, Wgb, nWD);
  cast_f32_to_bf16_kernel<<<(unsigned)(nWD / 256), 256, 0, stream>>>(Wo, Wob, nWD);

  // ---- projections: q (bf16), v (f32 for scan), g (bf16) ----
  dim3 gg(D_MODEL / BN, (unsigned)(ML / BM));     // (8, 256)
  gemm_xwT_kernel<1><<<gg, 256, 0, stream>>>(xb, Wqb, qb, (int)ML, D_MODEL, D_MODEL);
  gemm_xwT_kernel<0><<<gg, 256, 0, stream>>>(xb, Wvb, vst, (int)ML, D_MODEL, D_MODEL);
  gemm_xwT_kernel<1><<<gg, 256, 0, stream>>>(xb, Wgb, gb, (int)ML, D_MODEL, D_MODEL);

  // ---- chunked decay scan (3 phases; phase 3 fused into LN kernel) ----
  scan_chunk_kernel<<<(BATCH * D_MODEL * NCHUNK) / 256, 256, 0, stream>>>(
      vst, decay, carry);
  scan_carry_kernel<<<(BATCH * D_MODEL) / 256, 256, 0, stream>>>(carry, decay);

  // ---- fixup + q*state + LayerNorm + silu gate -> bf16 (reuse xb) ----
  ln_gate_kernel<<<(unsigned)ML, 256, 0, stream>>>(vst, qb, gb, carry, decay,
                                                   gamma, beta, xb);

  // ---- out = yg @ Wo^T (f32) ----
  gemm_xwT_kernel<0><<<gg, 256, 0, stream>>>(xb, Wob, (float*)d_out, (int)ML,
                                             D_MODEL, D_MODEL);
}
